// mLSTM_cell_23699629539944
// MI455X (gfx1250) — matrix operand order repaired and checked by reading, not compiled, not run on hardware
//
#include <hip/hip_runtime.h>
#include <hip/hip_bf16.h>
#include <math.h>

#define SEQL 2048
#define NH   8
#define HD   128
#define DIMF 1024

typedef __attribute__((ext_vector_type(16))) __bf16 v16bf;
typedef __attribute__((ext_vector_type(8)))  __bf16 v8bf;
typedef __attribute__((ext_vector_type(8)))  float  v8f;
typedef __attribute__((ext_vector_type(4)))  unsigned int u32x4;
typedef __attribute__((ext_vector_type(8)))  int i32x8;
typedef __attribute__((ext_vector_type(4)))  int i32x4;

union V16U { v16bf v; v8bf h[2]; };

__device__ __forceinline__ unsigned short f2bf(float x) {
    unsigned int u = __float_as_uint(x);
    u += 0x7fffu + ((u >> 16) & 1u);
    return (unsigned short)(u >> 16);
}

// Issue a TDM copy of one contiguous 8KB K-chunk (1024 x 8B elements) into LDS.
// 6-arg builtin form (clang-23 / therock-10.0 headers).
__device__ __forceinline__ void tdm_copy_8k(unsigned lds_addr, const void* gsrc) {
    unsigned long long ga = (unsigned long long)(uintptr_t)gsrc;
    u32x4 g0;
    g0[0] = 1u;                               // count=1 (valid), no gather
    g0[1] = lds_addr;                         // lds_addr (bytes, wave-relative)
    g0[2] = (unsigned)(ga & 0xffffffffu);     // global_addr[31:0]
    g0[3] = (unsigned)((ga >> 32) & 0x1ffffffu) | (2u << 30);  // addr[56:32] | type=2
    i32x8 g1;
    g1[0] = (int)(3u << 16);                  // data_size=3 (8B); mask/flags/pad = 0
    g1[1] = (int)(1024u << 16);               // tensor_dim0[15:0] = 1024
    g1[2] = (int)(1u << 16);                  // tensor_dim0 hi=0 | tensor_dim1 lo=1
    g1[3] = (int)(1024u << 16);               // tensor_dim1 hi=0 | tile_dim0=1024
    g1[4] = 0;                                // tile_dim1=0 (1-D), tile_dim2=0
    g1[5] = 1024;                             // tensor_dim0_stride lo
    g1[6] = 0;
    g1[7] = 0;
    i32x4 gz4;
    gz4[0] = 0; gz4[1] = 0; gz4[2] = 0; gz4[3] = 0;
    i32x8 gz8;
    gz8[0] = 0; gz8[1] = 0; gz8[2] = 0; gz8[3] = 0;
    gz8[4] = 0; gz8[5] = 0; gz8[6] = 0; gz8[7] = 0;
    __builtin_amdgcn_tensor_load_to_lds(g0, g1, gz4, gz4, gz8, 0);
}

// ---------------- prep: f32 [s][1024] -> per-head bf16 [h][s][128] ----------------
__global__ __launch_bounds__(256)
void prep_kernel(const float* __restrict__ q, const float* __restrict__ k,
                 const float* __restrict__ v,
                 unsigned short* __restrict__ qb, unsigned short* __restrict__ kb,
                 unsigned short* __restrict__ vb) {
    int idx = blockIdx.x * 256 + threadIdx.x;
    if (idx >= SEQL * DIMF) return;
    int s = idx >> 10;
    int d = idx & (DIMF - 1);
    int h = d >> 7;
    int e = d & (HD - 1);
    size_t dst = ((size_t)h * SEQL + s) * HD + e;
    qb[dst] = f2bf(q[idx]);
    kb[dst] = f2bf(k[idx]);
    vb[dst] = f2bf(v[idx]);
}

// ---------------- gates: itilde/ftilde [h][s] ----------------
__global__ __launch_bounds__(256)
void gates_kernel(const float* __restrict__ q, const float* __restrict__ k,
                  const float* __restrict__ v,
                  const float* __restrict__ Wi_w, const float* __restrict__ Wi_b,
                  const float* __restrict__ Wf_w, const float* __restrict__ Wf_b,
                  float* __restrict__ it, float* __restrict__ ft) {
    int w = threadIdx.x >> 5;             // wave -> token
    int lane = threadIdx.x & 31;
    int s = blockIdx.x * 8 + w;
    int h = lane >> 2;                    // 8 heads x 4 lanes
    int p = lane & 3;
    const float* Wi = Wi_w + h * 3 * DIMF;
    const float* Wf = Wf_w + h * 3 * DIMF;
    const float* qs = q + (size_t)s * DIMF;
    const float* ks = k + (size_t)s * DIMF;
    const float* vs = v + (size_t)s * DIMF;
    float ai = 0.f, af = 0.f;
    for (int d = p; d < 3 * DIMF; d += 4) {
        float g = (d < DIMF) ? qs[d] : (d < 2 * DIMF) ? ks[d - DIMF] : vs[d - 2 * DIMF];
        ai += g * Wi[d];
        af += g * Wf[d];
    }
    ai += __shfl_xor(ai, 1, 32); ai += __shfl_xor(ai, 2, 32);
    af += __shfl_xor(af, 1, 32); af += __shfl_xor(af, 2, 32);
    if (p == 0) {
        it[h * SEQL + s] = ai + Wi_b[h];
        ft[h * SEQL + s] = af + Wf_b[h];
    }
}

// ---------------- scan: u = itilde - cumsum(logsig(f)); rm = prefixmax(u); en = exp(-(csum+rm)) ----------------
__global__ void scan_kernel(const float* __restrict__ it, const float* __restrict__ ft,
                            float* __restrict__ u, float* __restrict__ rmo,
                            float* __restrict__ en) {
    int h = threadIdx.x;
    if (h >= NH) return;
    float cs = 0.f;
    float rmax = -3.0e38f;
    for (int s = 0; s < SEQL; ++s) {
        float f = ft[h * SEQL + s];
        float ls = fminf(f, 0.f) - log1pf(__expf(-fabsf(f)));   // stable logsigmoid
        cs += ls;
        float uu = it[h * SEQL + s] - cs;
        rmax = fmaxf(rmax, uu);
        u[h * SEQL + s]   = uu;
        rmo[h * SEQL + s] = rmax;
        en[h * SEQL + s]  = __expf(-(cs + rmax));               // exp(-max_d[i])
    }
}

// ---------------- main: decayed attention with WMMA bf16 + TDM double-buffered K ----------------
__global__ __launch_bounds__(128)
void mlstm_attn(const unsigned short* __restrict__ qb,
                const unsigned short* __restrict__ kb,
                const unsigned short* __restrict__ vb,
                const float* __restrict__ ug,
                const float* __restrict__ rmg,
                const float* __restrict__ eng,
                float* __restrict__ Ht) {
    __shared__ __align__(16) unsigned short kbuf[2][32 * HD]; // K chunk, row-major [j][d], double-buffered (TDM)
    __shared__ __align__(16) unsigned short vt[HD * 32];      // V chunk, transposed [d][j]
    __shared__ __align__(16) unsigned short cbuf[4][16 * 32]; // per-wave C re-layout scratch

    const float TAU_INV = 0.08838834764831845f;  // 1/sqrt(128)
    const int h    = blockIdx.y;
    const int i0   = blockIdx.x * 64;
    const int tid  = threadIdx.x;
    const int w    = tid >> 5;      // wave 0..3
    const int lane = tid & 31;
    const int nl   = lane & 15;     // column within 16-wide tile
    const int hi   = lane >> 4;     // half-wave

    const unsigned short* qh = qb + (size_t)h * SEQL * HD;
    const unsigned short* kh = kb + (size_t)h * SEQL * HD;
    const unsigned short* vh = vb + (size_t)h * SEQL * HD;
    const float* u  = ug  + h * SEQL;
    const float* rm = rmg + h * SEQL;
    const float* en = eng + h * SEQL;

    const int rowBase = i0 + w * 16;       // this wave's 16 rows

    // Q strip in WMMA A-layout: lane row m=nl, halves e: K = (e&7) + 8*hi + 16*(e>=8) + 32*c
    V16U a[4];
    {
        const unsigned short* qrow = qh + (size_t)(rowBase + nl) * HD;
        #pragma unroll
        for (int c = 0; c < 4; ++c) {
            a[c].h[0] = *reinterpret_cast<const v8bf*>(qrow + c * 32 + 8 * hi);
            a[c].h[1] = *reinterpret_cast<const v8bf*>(qrow + c * 32 + 16 + 8 * hi);
        }
    }

    float rm_i[8], en_i[8], rs[8];
    #pragma unroll
    for (int r = 0; r < 8; ++r) {
        int i = rowBase + r + 8 * hi;
        rm_i[r] = rm[i];
        en_i[r] = en[i];
        rs[r] = 0.f;
    }

    v8f acc[8];
    #pragma unroll
    for (int ot = 0; ot < 8; ++ot)
        #pragma unroll
        for (int e = 0; e < 8; ++e) acc[ot][e] = 0.f;

    const int iMaxBlock = i0 + 63;
    const int iMaxWave  = rowBase + 15;

    const unsigned lds_k0 = (unsigned)(uintptr_t)&kbuf[0][0];
    const unsigned lds_k1 = (unsigned)(uintptr_t)&kbuf[1][0];

    // Prime the pipeline: TDM-load chunk 0 into buffer 0 (wave 0 only; EXEC-independent op).
    if (w == 0) {
        tdm_copy_8k(lds_k0, kh);
    }

    for (int jb = 0; jb <= iMaxBlock; jb += 32) {
        const int cur = (jb >> 5) & 1;
        __syncthreads();   // previous iteration's readers of vt / kbuf[cur^1] are done
        {
            // V chunk: transpose-stage into LDS (VALU path; TDM cannot swap the contiguous axis)
            int r = tid >> 2;              // 0..31  (token within chunk)
            int c = (tid & 3) * 32;        // 0,32,64,96 (feature)
            const unsigned short* vsrc = vh + (size_t)(jb + r) * HD + c;
            #pragma unroll
            for (int e = 0; e < 32; ++e) vt[(c + e) * 32 + r] = vsrc[e];
            if (jb + 32 <= iMaxBlock) {    // hint next V chunk into cache
                __builtin_prefetch(vh + (size_t)(jb + 32 + r) * HD + c, 0, 1);
            }
        }
        // Wave 0: kick off DMA for the NEXT K chunk, then wait until the CURRENT one landed.
        if (w == 0) {
            if (jb + 32 <= iMaxBlock) {
                tdm_copy_8k(cur ? lds_k0 : lds_k1,
                            kh + (size_t)(jb + 32) * HD);
                __builtin_amdgcn_s_wait_tensorcnt(1);  // in-order: older (current) chunk done
            } else {
                __builtin_amdgcn_s_wait_tensorcnt(0);
            }
        }
        __syncthreads();   // publish vt + kbuf[cur] to all waves
        if (jb > iMaxWave) continue;       // wave-uniform; barrier count stays uniform

        const unsigned short* kt = &kbuf[cur][0];

        // ---- GEMM1: S(16x32) = Q(16x128) . K^T, two 16-wide n-tiles ----
        v8f sacc[2];
        #pragma unroll
        for (int nt = 0; nt < 2; ++nt)
            #pragma unroll
            for (int e = 0; e < 8; ++e) sacc[nt][e] = 0.f;

        #pragma unroll
        for (int nt = 0; nt < 2; ++nt) {
            #pragma unroll
            for (int c = 0; c < 4; ++c) {
                V16U bu;
                bu.h[0] = *reinterpret_cast<const v8bf*>(kt + (nt * 16 + nl) * HD + c * 32 + 8 * hi);
                bu.h[1] = *reinterpret_cast<const v8bf*>(kt + (nt * 16 + nl) * HD + c * 32 + 16 + 8 * hi);
                v16bf b = bu.v;
                sacc[nt] = __builtin_amdgcn_wmma_f32_16x16x32_bf16(
                    false, a[c].v, false, b, (short)0, sacc[nt], false, false);
            }
        }

        // ---- decay, causal mask, rowsum; write C tile to LDS as bf16 ----
        #pragma unroll
        for (int nt = 0; nt < 2; ++nt) {
            int j = jb + nt * 16 + nl;
            float uj = u[j];
            #pragma unroll
            for (int r = 0; r < 8; ++r) {
                int i = rowBase + r + 8 * hi;
                float val = (j <= i) ? sacc[nt][r] * TAU_INV * __expf(uj - rm_i[r]) : 0.f;
                rs[r] += val;
                cbuf[w][(r + 8 * hi) * 32 + nt * 16 + nl] = f2bf(val);
            }
        }

        // ---- re-read C in A-layout (intra-wave LDS ops are in order) ----
        V16U cA;
        cA.h[0] = *reinterpret_cast<const v8bf*>(&cbuf[w][nl * 32 + 8 * hi]);
        cA.h[1] = *reinterpret_cast<const v8bf*>(&cbuf[w][nl * 32 + 16 + 8 * hi]);

        // ---- GEMM2: Ht(16x128) += C(16x32) . V(32x128) ----
        #pragma unroll
        for (int ot = 0; ot < 8; ++ot) {
            V16U b2u;
            b2u.h[0] = *reinterpret_cast<const v8bf*>(vt + (ot * 16 + nl) * 32 + 8 * hi);
            b2u.h[1] = *reinterpret_cast<const v8bf*>(vt + (ot * 16 + nl) * 32 + 16 + 8 * hi);
            v16bf b2 = b2u.v;
            acc[ot] = __builtin_amdgcn_wmma_f32_16x16x32_bf16(
                false, cA.v, false, b2, (short)0, acc[ot], false, false);
        }
    }

    // ---- rowsum reduce across the 16 lanes of each half-wave ----
    #pragma unroll
    for (int r = 0; r < 8; ++r) {
        float s = rs[r];
        s += __shfl_xor(s, 1, 32);
        s += __shfl_xor(s, 2, 32);
        s += __shfl_xor(s, 4, 32);
        s += __shfl_xor(s, 8, 32);
        rs[r] = s;
    }

    // ---- scale by 1/(maxit + 1e-6), store pre-norm H [s][h*128+d] ----
    #pragma unroll
    for (int r = 0; r < 8; ++r) {
        int i = rowBase + r + 8 * hi;
        float maxit = fmaxf(fabsf(rs[r]), en_i[r]);
        float inv = 1.f / (maxit + 1e-6f);
        #pragma unroll
        for (int ot = 0; ot < 8; ++ot) {
            int d = ot * 16 + nl;
            Ht[(size_t)i * DIMF + h * HD + d] = acc[ot][r] * inv;
        }
    }
}

// ---------------- GroupNorm (in-place on d_out) ----------------
__global__ __launch_bounds__(256)
void gnorm_kernel(float* __restrict__ out, const float* __restrict__ gw,
                  const float* __restrict__ gb) {
    int s = blockIdx.x;
    int w = threadIdx.x >> 5;     // head
    int lane = threadIdx.x & 31;
    float* x = out + (size_t)s * DIMF + w * HD;
    float4 xv = reinterpret_cast<const float4*>(x)[lane];
    float sum = xv.x + xv.y + xv.z + xv.w;
    float sq  = xv.x * xv.x + xv.y * xv.y + xv.z * xv.z + xv.w * xv.w;
    #pragma unroll
    for (int m = 1; m < 32; m <<= 1) {
        sum += __shfl_xor(sum, m, 32);
        sq  += __shfl_xor(sq,  m, 32);
    }
    float mean = sum * (1.f / HD);
    float var  = sq * (1.f / HD) - mean * mean;
    float rstd = rsqrtf(var + 1e-5f);
    int d = w * HD + lane * 4;
    float4 o;
    o.x = (xv.x - mean) * rstd * gw[d + 0] + gb[d + 0];
    o.y = (xv.y - mean) * rstd * gw[d + 1] + gb[d + 1];
    o.z = (xv.z - mean) * rstd * gw[d + 2] + gb[d + 2];
    o.w = (xv.w - mean) * rstd * gw[d + 3] + gb[d + 3];
    reinterpret_cast<float4*>(x)[lane] = o;
}

// ---------------- launch ----------------
extern "C" void kernel_launch(void* const* d_in, const int* in_sizes, int n_in,
                              void* d_out, int out_size, void* d_ws, size_t ws_size,
                              hipStream_t stream) {
    const float* q    = (const float*)d_in[0];
    const float* k    = (const float*)d_in[1];
    const float* v    = (const float*)d_in[2];
    const float* Wi_w = (const float*)d_in[3];
    const float* Wi_b = (const float*)d_in[4];
    const float* Wf_w = (const float*)d_in[5];
    const float* Wf_b = (const float*)d_in[6];
    const float* gn_w = (const float*)d_in[7];
    const float* gn_b = (const float*)d_in[8];
    float* out = (float*)d_out;

    // workspace layout
    unsigned short* qb = (unsigned short*)d_ws;
    unsigned short* kb = qb + (size_t)SEQL * DIMF;
    unsigned short* vb = kb + (size_t)SEQL * DIMF;
    float* it  = (float*)(vb + (size_t)SEQL * DIMF);
    float* ft  = it  + NH * SEQL;
    float* u   = ft  + NH * SEQL;
    float* rmx = u   + NH * SEQL;
    float* en  = rmx + NH * SEQL;

    prep_kernel<<<(SEQL * DIMF + 255) / 256, 256, 0, stream>>>(q, k, v, qb, kb, vb);
    gates_kernel<<<SEQL / 8, 256, 0, stream>>>(q, k, v, Wi_w, Wi_b, Wf_w, Wf_b, it, ft);
    scan_kernel<<<1, 32, 0, stream>>>(it, ft, u, rmx, en);
    dim3 grid(SEQL / 64, NH);
    mlstm_attn<<<grid, 128, 0, stream>>>(qb, kb, vb, u, rmx, en, out);
    gnorm_kernel<<<SEQL, 256, 0, stream>>>(out, gn_w, gn_b);
}
